// AECModel_5437428596954
// MI455X (gfx1250) — matrix operand order repaired and checked by reading, not compiled, not run on hardware
//
#include <hip/hip_runtime.h>

// ---------------------------------------------------------------------------
// AEC model (17x pointwise 32ch convs + 3 GRU scans) for MI455X / gfx1250.
//
// Strategy (see analysis):
//  * all matmuls via v_wmma_f32_16x16x32_f16 (wave32, 16x16 tiles, K=32)
//  * activations stored f16 in workspace (bandwidth-bound convs: ~1.1 GB
//    total traffic -> ~50us at 23.3 TB/s)
//  * GRU: gi = x@Wih.T + bih + bhh precomputed with WMMA and stored in the
//    *WMMA C-register layout* so the latency-bound scan loads it with one
//    b128 per gate tile; recurrent weights held in VGPRs across all 4096
//    steps; r-gate never computed (reference discards it); h transposed via
//    1KB LDS per wave (same-wave DS ops are in-order on CDNA5).
//  * workspace layout (needs ~129 MB):
//      [0, 256KB)        packed f16 weight tiles (77 tiles x 1KB)
//      buf0, buf1        f16 [B*T, 32]   (16 MB each)
//      cat               f16 [B*T, 64]   (32 MB)
//      giA, giB          f16 swizzled [T][4][4][32][8] (32 MB each)
// ---------------------------------------------------------------------------

typedef _Float16 h16;
typedef __attribute__((ext_vector_type(16))) _Float16 v16h;
typedef __attribute__((ext_vector_type(8)))  _Float16 v8h;
typedef __attribute__((ext_vector_type(8)))  float    v8f;
typedef __attribute__((ext_vector_type(4)))  float    v4f;

__device__ __forceinline__ unsigned pack2h(float lo, float hi) {
  union { h16 h[2]; unsigned u; } u;
  u.h[0] = (h16)lo; u.h[1] = (h16)hi; return u.u;
}
__device__ __forceinline__ float sigmoidf_(float x) {
  return 1.f / (1.f + __expf(-x));
}
__device__ __forceinline__ float tanhf_(float x) {
  x = fminf(fmaxf(x, -15.f), 15.f);
  float e = __expf(-2.f * x);
  return (1.f - e) / (1.f + e);
}
__device__ __forceinline__ v8f wmma16(v16h a, v16h b, v8f c) {
  // D = A(16x32 f16) * B(32x16 f16) + C(16x16 f32)
  return __builtin_amdgcn_wmma_f32_16x16x32_f16(false, a, false, b,
                                                (short)0, c, false, false);
}

// ---------------------------------------------------------------------------
// Pack weight matrix W [Cout, Cin] (f32, row-major) into WMMA B-tile layout:
// B = W^T (K=Cin, N=Cout). Per 32x16 tile: lane L, dword j holds packed pair
// B[K = (L/16)*8 + 2j + (j>=4 ? 8 : 0) (+kt*32), N = L%16 (+nt*16)]. Zero-padded OOB.
// Tile index = mat*(tilesK*tilesN) + kt*tilesN + nt; 256 dwords per tile.
// ---------------------------------------------------------------------------
__global__ void pack_b(const float* __restrict__ W, int Cout, int Cin,
                       int nMats, unsigned* __restrict__ dst) {
  int tilesN = (Cout + 15) >> 4, tilesK = (Cin + 31) >> 5;
  int tpm = tilesN * tilesK;
  int tile = blockIdx.x;
  int mat = tile / tpm, rem = tile % tpm;
  int kt = rem / tilesN, nt = rem % tilesN;
  const float* Wm = W + (size_t)mat * Cout * Cin;
  int lane = threadIdx.x;
  int halfid = lane >> 4, n = nt * 16 + (lane & 15);
  unsigned* o = dst + (size_t)tile * 256 + lane * 8;
#pragma unroll
  for (int j = 0; j < 8; ++j) {
    int k = kt * 32 + halfid * 8 + 2 * j + (j >= 4 ? 8 : 0);
    float lo = (n < Cout && k     < Cin) ? Wm[(size_t)n * Cin + k]     : 0.f;
    float hi = (n < Cout && k + 1 < Cin) ? Wm[(size_t)n * Cin + k + 1] : 0.f;
    o[j] = pack2h(lo, hi);
  }
  (void)nMats;
}

// Load one A-tile (16 rows x 32 K) in WMMA 16-bit A layout for this lane.
// Lane L: elems [0..7] = K = halfid*8 + j (+kt*32), [8..15] = K+16.
__device__ __forceinline__ v16h load_a_tile(const void* in, int inF32, int cin,
                                            long row, int kt, int halfid) {
  v16h av;
  int k0 = kt * 32 + halfid * 8, k1 = k0 + 16;
  if (inF32) {
    const float* p = (const float*)in + row * (long)cin;
    if (k1 + 8 <= cin) {
      v4f x0 = *(const v4f*)(p + k0), x1 = *(const v4f*)(p + k0 + 4);
      v4f y0 = *(const v4f*)(p + k1), y1 = *(const v4f*)(p + k1 + 4);
#pragma unroll
      for (int j = 0; j < 4; ++j) {
        av[j] = (h16)x0[j]; av[4 + j] = (h16)x1[j];
        av[8 + j] = (h16)y0[j]; av[12 + j] = (h16)y1[j];
      }
    } else {
#pragma unroll
      for (int j = 0; j < 8; ++j) av[j]     = (k0 + j < cin) ? (h16)p[k0 + j] : (h16)0.f;
#pragma unroll
      for (int j = 0; j < 8; ++j) av[8 + j] = (k1 + j < cin) ? (h16)p[k1 + j] : (h16)0.f;
    }
  } else {
    const h16* p = (const h16*)in + row * (long)cin;
    v8h x = *(const v8h*)(p + k0);
    v8h y = *(const v8h*)(p + k1);
#pragma unroll
    for (int j = 0; j < 8; ++j) { av[j] = x[j]; av[8 + j] = y[j]; }
  }
  return av;
}

// ---------------------------------------------------------------------------
// Generic pointwise conv: out[r,:] = act(in[r,:] @ W^T + b).
// 128 threads = 4 waves; each wave owns one 16-row tile, loops over N tiles.
// act: 0=none, 1=relu, 2=sigmoid.  out is f16 (outF16=1) or f32.
// ---------------------------------------------------------------------------
__global__ void conv_wmma(const void* __restrict__ in, int inF32, int cin,
                          void* __restrict__ out, int outF16, int ostride,
                          const unsigned* __restrict__ wq, int tileBase,
                          int kTiles, int nTiles,
                          const float* __restrict__ bias, int cout, int act,
                          int R) {
  int lane = threadIdx.x & 31;
  int wv = threadIdx.x >> 5;
  int rt = blockIdx.x * 4 + wv;
  if (rt * 16 >= R) return;  // uniform per wave: safe for WMMA
  int halfid = lane >> 4, col = lane & 15;
  long arow = (long)rt * 16 + col;

  v16h a0 = load_a_tile(in, inF32, cin, arow, 0, halfid);
  v16h a1;
  if (kTiles > 1) a1 = load_a_tile(in, inF32, cin, arow, 1, halfid);

  for (int ct = 0; ct < nTiles; ++ct) {
    int ocol = ct * 16 + col;
    float bv = (ocol < cout) ? bias[ocol] : 0.f;
    v8f c;
#pragma unroll
    for (int j = 0; j < 8; ++j) c[j] = bv;
    c = wmma16(a0, *(const v16h*)(wq + (size_t)(tileBase + ct) * 256 + lane * 8), c);
    if (kTiles > 1)
      c = wmma16(a1, *(const v16h*)(wq + (size_t)(tileBase + nTiles + ct) * 256 + lane * 8), c);
#pragma unroll
    for (int j = 0; j < 8; ++j) {
      float v = c[j];
      if (act == 1)      v = fmaxf(v, 0.f);
      else if (act == 2) v = sigmoidf_(v);
      c[j] = v;
    }
    if (ocol < cout) {
#pragma unroll
      for (int r = 0; r < 8; ++r) {
        long m = (long)rt * 16 + r + 8 * halfid;  // D layout: M = r + 8*(lane/16)
        if (outF16) ((h16*)out)[m * ostride + ocol] = (h16)c[r];
        else        ((float*)out)[m * ostride + ocol] = c[r];
      }
    }
  }
}

// ---------------------------------------------------------------------------
// gi = h @ Wih^T + bih + bhh for gate cols 32..95 (z,n only; r is unused by
// the reference cell). Output stored pre-swizzled in WMMA C-register layout:
//   gi[t][bt][j][lane] = v8h   (one b128 per tile per lane in the scan)
// grid = T blocks x 128 threads; wave w handles batch tile bt = w.
// ---------------------------------------------------------------------------
__global__ void gi_wmma(const h16* __restrict__ in,  // [B*T, 32] f16, natural
                        const unsigned* __restrict__ wq, int wihBase,
                        const float* __restrict__ bih,
                        const float* __restrict__ bhh,
                        v8h* __restrict__ giOut, int T) {
  int lane = threadIdx.x & 31;
  int bt = threadIdx.x >> 5;
  int t = blockIdx.x;
  int halfid = lane >> 4, col = lane & 15;
  int b = bt * 16 + col;  // A row = batch index (stride T*32 between rows)
  const h16* p = in + ((size_t)b * T + t) * 32;
  v8h x = *(const v8h*)(p + halfid * 8);
  v8h y = *(const v8h*)(p + 16 + halfid * 8);
  v16h a;
#pragma unroll
  for (int j = 0; j < 8; ++j) { a[j] = x[j]; a[8 + j] = y[j]; }

#pragma unroll
  for (int j = 0; j < 4; ++j) {  // tiles over gate cols 32+16j .. +15
    int gcol = 32 + j * 16 + col;
    float bv = bih[gcol] + bhh[gcol];
    v8f c;
#pragma unroll
    for (int r = 0; r < 8; ++r) c[r] = bv;
    c = wmma16(a, *(const v16h*)(wq + (size_t)(wihBase + 2 + j) * 256 + lane * 8), c);
    v8h g;
#pragma unroll
    for (int r = 0; r < 8; ++r) g[r] = (h16)c[r];
    giOut[(((size_t)t * 4 + bt) * 4 + j) * 32 + lane] = g;
  }
}

// ---------------------------------------------------------------------------
// Sequential GRU scan. One wave per block; block handles 16 batch rows.
// grid = 4 (uni) or 8 (bidirectional: blocks 0-3 fwd, 4-7 rev).
// Per step: 4 x WMMA (z0,z1,n0,n1) with gi preloaded into C; h kept in
// D-layout VGPRs; f16 copy transposed through 1KB LDS for the next A.
// ---------------------------------------------------------------------------
__global__ void gru_scan(const unsigned* __restrict__ wq,
                         const v8h* __restrict__ giA, int whhA, int colOffA, int dirA,
                         const v8h* __restrict__ giB, int whhB, int colOffB, int dirB,
                         h16* __restrict__ out, int ostride, int T) {
  int lane = threadIdx.x;
  int dirIdx = blockIdx.x >> 2;
  int bt = blockIdx.x & 3;
  const v8h* gi = dirIdx ? giB : giA;
  int whhBase = dirIdx ? whhB : whhA;
  int colOff  = dirIdx ? colOffB : colOffA;
  int dir     = dirIdx ? dirB : dirA;
  int halfid = lane >> 4, col = lane & 15;

  __shared__ h16 hl[2][16][16];  // [K/16][row][K%16]
  for (int i = lane; i < 512; i += 32) ((h16*)hl)[i] = (h16)0.f;

  // Recurrent weight tiles (z0,z1,n0,n1) pinned in VGPRs for all T steps.
  v16h w[4];
#pragma unroll
  for (int j = 0; j < 4; ++j)
    w[j] = *(const v16h*)(wq + (size_t)(whhBase + 2 + j) * 256 + lane * 8);

  v8f h0, h1;  // current hidden state, D-register layout (cols 0-15 / 16-31)
#pragma unroll
  for (int r = 0; r < 8; ++r) { h0[r] = 0.f; h1[r] = 0.f; }

  int t = (dir > 0) ? 0 : T - 1;
  for (int s = 0; s < T; ++s, t += dir) {
    // A (16x32 f16) from LDS-transposed h
    v8h x = *(const v8h*)&hl[0][col][halfid * 8];
    v8h y = *(const v8h*)&hl[1][col][halfid * 8];
    v16h a;
#pragma unroll
    for (int j = 0; j < 8; ++j) { a[j] = x[j]; a[8 + j] = y[j]; }

    size_t gbase = (((size_t)t * 4 + bt) * 4) * 32 + lane;
    v8f acc[4];
#pragma unroll
    for (int j = 0; j < 4; ++j) {
      v8h g = gi[gbase + (size_t)j * 32];
      v8f c;
#pragma unroll
      for (int r = 0; r < 8; ++r) c[r] = (float)g[r];
      acc[j] = wmma16(a, w[j], c);
    }
    int tn = t + dir;
    if (tn >= 0 && tn < T)
      __builtin_prefetch((const void*)(gi + (((size_t)tn * 4 + bt) * 4) * 32 + lane), 0, 0);

#pragma unroll
    for (int r = 0; r < 8; ++r) {
      float z = sigmoidf_(acc[0][r]);
      float n = tanhf_(acc[2][r]);
      h0[r] = (1.f - z) * n + z * h0[r];
      z = sigmoidf_(acc[1][r]);
      n = tanhf_(acc[3][r]);
      h1[r] = (1.f - z) * n + z * h1[r];
    }

    // emit outputs (natural layout) + refresh LDS transpose of h
#pragma unroll
    for (int r = 0; r < 8; ++r) {
      int m = r + 8 * halfid;
      long b = (long)bt * 16 + m;
      out[(b * T + t) * ostride + colOff + col]      = (h16)h0[r];
      out[(b * T + t) * ostride + colOff + 16 + col] = (h16)h1[r];
      hl[0][m][col] = (h16)h0[r];
      hl[1][m][col] = (h16)h1[r];
    }
    // Same-wave DS ops are in-order; wait so next-iter loads see the stores.
    asm volatile("s_wait_dscnt 0x0" ::: "memory");
  }
}

// ---------------------------------------------------------------------------
extern "C" void kernel_launch(void* const* d_in, const int* in_sizes, int n_in,
                              void* d_out, int out_size, void* d_ws,
                              size_t ws_size, hipStream_t stream) {
  (void)in_sizes; (void)n_in; (void)out_size; (void)ws_size;
  const float* x     = (const float*)d_in[0];
  const float* w_in  = (const float*)d_in[1];
  const float* b_in  = (const float*)d_in[2];
  const float* W32   = (const float*)d_in[3];
  const float* B32   = (const float*)d_in[4];
  const float* w_up1 = (const float*)d_in[5];
  const float* b_up1 = (const float*)d_in[6];
  const float* w_out = (const float*)d_in[7];
  const float* b_out = (const float*)d_in[8];
  const float* wih1  = (const float*)d_in[9];
  const float* whh1  = (const float*)d_in[10];
  const float* bih1  = (const float*)d_in[11];
  const float* bhh1  = (const float*)d_in[12];
  const float* wih2  = (const float*)d_in[13];
  const float* whh2  = (const float*)d_in[14];
  const float* bih2  = (const float*)d_in[15];
  const float* bhh2  = (const float*)d_in[16];
  const float* wih2r = (const float*)d_in[17];
  const float* whh2r = (const float*)d_in[18];
  const float* bih2r = (const float*)d_in[19];
  const float* bhh2r = (const float*)d_in[20];

  const int B = 64, T = 4096, R = B * T;

  char* ws = (char*)d_ws;
  unsigned* wq = (unsigned*)ws;                 // packed weight tiles
  size_t o = (size_t)1 << 18;                   // 256 KB reserved for weights
  h16* buf0 = (h16*)(ws + o); o += (size_t)R * 32 * 2;
  h16* buf1 = (h16*)(ws + o); o += (size_t)R * 32 * 2;
  h16* cat  = (h16*)(ws + o); o += (size_t)R * 64 * 2;
  v8h* giA  = (v8h*)(ws + o); o += (size_t)T * 4 * 4 * 32 * 16;
  v8h* giB  = (v8h*)(ws + o);

  enum { OFF_WIN = 0, OFF_W32 = 2, OFF_UP1 = 36, OFF_WOUT = 40,
         OFF_WIH1 = 41, OFF_WHH1 = 47, OFF_WIH2 = 53, OFF_WHH2 = 59,
         OFF_WIH2R = 65, OFF_WHH2R = 71 };  // tile offsets (1 tile = 256 dw)

  // --- weight packing (tiny) ---
  pack_b<<<2, 32, 0, stream>>>(w_in, 32, 8, 1, wq + OFF_WIN * 256);
  pack_b<<<34, 32, 0, stream>>>(W32, 32, 32, 17, wq + OFF_W32 * 256);
  pack_b<<<4, 32, 0, stream>>>(w_up1, 32, 64, 1, wq + OFF_UP1 * 256);
  pack_b<<<1, 32, 0, stream>>>(w_out, 2, 32, 1, wq + OFF_WOUT * 256);
  pack_b<<<6, 32, 0, stream>>>(wih1, 96, 32, 1, wq + OFF_WIH1 * 256);
  pack_b<<<6, 32, 0, stream>>>(whh1, 96, 32, 1, wq + OFF_WHH1 * 256);
  pack_b<<<6, 32, 0, stream>>>(wih2, 96, 32, 1, wq + OFF_WIH2 * 256);
  pack_b<<<6, 32, 0, stream>>>(whh2, 96, 32, 1, wq + OFF_WHH2 * 256);
  pack_b<<<6, 32, 0, stream>>>(wih2r, 96, 32, 1, wq + OFF_WIH2R * 256);
  pack_b<<<6, 32, 0, stream>>>(whh2r, 96, 32, 1, wq + OFF_WHH2R * 256);

  dim3 cgrid(R / 64), cblk(128);
  // input conv 8->32 (f32 in, f16 out)
  conv_wmma<<<cgrid, cblk, 0, stream>>>(x, 1, 8, buf0, 1, 32, wq, OFF_WIN,
                                        1, 2, b_in, 32, 1, R);
  h16* cur = buf0; h16* nxt = buf1;
  for (int i = 0; i < 10; ++i) {  // down1-5, feat1-5
    conv_wmma<<<cgrid, cblk, 0, stream>>>(cur, 0, 32, nxt, 1, 32, wq,
                                          OFF_W32 + i * 2, 1, 2,
                                          B32 + i * 32, 32, 1, R);
    h16* tmp = cur; cur = nxt; nxt = tmp;
  }
  // gru1
  gi_wmma<<<T, 128, 0, stream>>>(cur, wq, OFF_WIH1, bih1, bhh1, giA, T);
  gru_scan<<<4, 32, 0, stream>>>(wq, giA, OFF_WHH1, 0, 1,
                                 giA, OFF_WHH1, 0, 1, nxt, 32, T);
  { h16* tmp = cur; cur = nxt; nxt = tmp; }
  for (int i = 10; i < 13; ++i) {  // dec1-3
    conv_wmma<<<cgrid, cblk, 0, stream>>>(cur, 0, 32, nxt, 1, 32, wq,
                                          OFF_W32 + i * 2, 1, 2,
                                          B32 + i * 32, 32, 1, R);
    h16* tmp = cur; cur = nxt; nxt = tmp;
  }
  // gru2 (bidirectional, scans run concurrently in one launch)
  gi_wmma<<<T, 128, 0, stream>>>(cur, wq, OFF_WIH2, bih2, bhh2, giA, T);
  gi_wmma<<<T, 128, 0, stream>>>(cur, wq, OFF_WIH2R, bih2r, bhh2r, giB, T);
  gru_scan<<<8, 32, 0, stream>>>(wq, giA, OFF_WHH2, 0, 1,
                                 giB, OFF_WHH2R, 32, -1, cat, 64, T);
  // up1: 64->32
  conv_wmma<<<cgrid, cblk, 0, stream>>>(cat, 0, 64, buf0, 1, 32, wq, OFF_UP1,
                                        2, 2, b_up1, 32, 1, R);
  cur = buf0; nxt = buf1;
  for (int i = 13; i < 17; ++i) {  // up2-5
    conv_wmma<<<cgrid, cblk, 0, stream>>>(cur, 0, 32, nxt, 1, 32, wq,
                                          OFF_W32 + i * 2, 1, 2,
                                          B32 + i * 32, 32, 1, R);
    h16* tmp = cur; cur = nxt; nxt = tmp;
  }
  // output conv 32->2 + sigmoid, f32 to d_out
  conv_wmma<<<cgrid, cblk, 0, stream>>>(cur, 0, 32, d_out, 0, 2, wq, OFF_WOUT,
                                        1, 1, b_out, 2, 2, R);
}
